// Q_DenseUnit_45432164057486
// MI455X (gfx1250) — 8-bit matrix operand order repaired and checked by reading, compiled, not run on hardware
//
#include <hip/hip_runtime.h>
#include <hip/hip_bf16.h>

typedef __attribute__((ext_vector_type(8))) int v8i;
typedef __attribute__((ext_vector_type(4))) int v4i;
typedef __attribute__((ext_vector_type(2))) int v2i;
typedef __attribute__((ext_vector_type(4))) unsigned int u32x4;

#define EPSF 1e-5f

#define OFF_WBN  0
#define OFF_BBN  1024
#define OFF_SBN  2048
#define OFF_W1   4096
#define OFF_B1   36864
#define OFF_SC1  37888
#define OFF_W2   38912
#define OFF_SC2  75776
#define OFF_Q1   76800
#define OFF_Q2   (76800 + 25690112)

__device__ __forceinline__ float quant_clip(float x) {
    return fminf(fmaxf(rintf(x), -128.0f), 127.0f);
}

__device__ __forceinline__ void tdm_load_1d(unsigned lds_off, const void* gptr,
                                            unsigned nbytes) {
#if __has_builtin(__builtin_amdgcn_tensor_load_to_lds)
    unsigned long long ga = (unsigned long long)(uintptr_t)gptr;
    u32x4 g0;
    g0[0] = 1u;
    g0[1] = lds_off;
    g0[2] = (unsigned)(ga & 0xffffffffu);
    g0[3] = (unsigned)((ga >> 32) & 0x01ffffffu)
          | (2u << 30);
    v8i g1;
    g1[0] = 0;
    g1[1] = (int)((nbytes & 0xffffu) << 16);
    g1[2] = (int)(((nbytes >> 16) & 0xffffu) | (1u << 16));
    g1[3] = (int)((nbytes & 0xffffu) << 16);
    g1[4] = 0;
    g1[5] = (int)nbytes;
    g1[6] = 0;
    g1[7] = 0;
    v4i z4; z4[0] = 0; z4[1] = 0; z4[2] = 0; z4[3] = 0;
    v8i z8; z8[0] = 0; z8[1] = 0; z8[2] = 0; z8[3] = 0;
            z8[4] = 0; z8[5] = 0; z8[6] = 0; z8[7] = 0;
    __builtin_amdgcn_tensor_load_to_lds(g0, g1, z4, z4, z8, 0);
#else
    (void)lds_off; (void)gptr; (void)nbytes;
#endif
}

__device__ __forceinline__ void tdm_wait() {
#if __has_builtin(__builtin_amdgcn_s_wait_tensorcnt)
    __builtin_amdgcn_s_wait_tensorcnt(0);
#endif
}

__global__ __launch_bounds__(256) void prep_bn(
    const float* __restrict__ g, const float* __restrict__ be,
    const float* __restrict__ mu, const float* __restrict__ va,
    const float* __restrict__ s_in_p, const float* __restrict__ s_out_p,
    float* __restrict__ wint, float* __restrict__ bint,
    float* __restrict__ sbn, float* __restrict__ out_tail) {
    __shared__ float red[256];
    int c = threadIdx.x;
    float w = g[c] * rsqrtf(va[c] + EPSF);
    float b = be[c] - mu[c] * w;
    red[c] = fabsf(w);
    __syncthreads();
    for (int s = 128; s > 0; s >>= 1) {
        if (c < s) red[c] = fmaxf(red[c], red[c + s]);
        __syncthreads();
    }
    float ws = red[0] / 127.0f;
    float s_in = s_in_p[0];
    wint[c] = quant_clip(w / ws);
    bint[c] = rintf(b / (ws * s_in));
    if (c == 0) {
        sbn[0] = ws * s_in;
        out_tail[0] = s_out_p[0];
    }
}

__global__ __launch_bounds__(256) void prep_c1(
    const float* __restrict__ w1f, const float* __restrict__ g2,
    const float* __restrict__ be2, const float* __restrict__ mu2,
    const float* __restrict__ va2, const float* __restrict__ s1p,
    signed char* __restrict__ w1q, float* __restrict__ b1f, float* __restrict__ sc1) {
    __shared__ float red[256];
    int m = blockIdx.x, c = threadIdx.x;
    float f  = g2[m] * rsqrtf(va2[m] + EPSF);
    float wf = w1f[m * 256 + c] * f;
    red[c] = fabsf(wf);
    __syncthreads();
    for (int s = 128; s > 0; s >>= 1) {
        if (c < s) red[c] = fmaxf(red[c], red[c + s]);
        __syncthreads();
    }
    float ws = red[0] / 127.0f;
    w1q[m * 256 + c] = (signed char)(int)quant_clip(wf / ws);
    if (c == 0) {
        float s1 = s1p[0];
        float bf = be2[m] - mu2[m] * f;
        b1f[m] = rintf(bf / (ws * s1));
        sc1[m] = ws * s1;
    }
}

__global__ __launch_bounds__(128) void prep_c2(
    const float* __restrict__ w2f, const float* __restrict__ s2p,
    signed char* __restrict__ w2q, float* __restrict__ sc2) {
    __shared__ float red[128];
    int m = blockIdx.x, c = threadIdx.x;
    float v[9];
    float mx = 0.0f;
    #pragma unroll
    for (int t = 0; t < 9; t++) {
        v[t] = w2f[(m * 128 + c) * 9 + t];
        mx = fmaxf(mx, fabsf(v[t]));
    }
    red[c] = mx;
    __syncthreads();
    for (int s = 64; s > 0; s >>= 1) {
        if (c < s) red[c] = fmaxf(red[c], red[c + s]);
        __syncthreads();
    }
    float ws = red[0] / 127.0f;
    #pragma unroll
    for (int t = 0; t < 9; t++)
        w2q[m * 1152 + t * 128 + c] = (signed char)(int)quant_clip(v[t] / ws);
    if (c == 0) sc2[m] = ws * s2p[0];
}

__global__ __launch_bounds__(256) void stage1(
    const float* __restrict__ batch, const float* __restrict__ s_in_p,
    const float* __restrict__ s1p, const float* __restrict__ s_out_p,
    const float* __restrict__ wint, const float* __restrict__ bint,
    const float* __restrict__ sbn_p, signed char* __restrict__ q1,
    float* __restrict__ out) {
    long long t = (long long)blockIdx.x * 256 + threadIdx.x;
    int n  = (int)(t >> 6);
    int c  = (int)(t & 63) * 4;
    int b  = n / 3136;
    int hw = n - b * 3136;
    float s_in = s_in_p[0], s1 = s1p[0], s_out = s_out_p[0], sbn = sbn_p[0];
    const float* bp = batch + (size_t)b * 256 * 3136 + (size_t)c * 3136 + hw;
    float*       op = out   + (size_t)b * 288 * 3136 + (size_t)c * 3136 + hw;
    unsigned pack = 0;
    #pragma unroll
    for (int i = 0; i < 4; i++) {
        float xv = bp[(size_t)i * 3136];
        op[(size_t)i * 3136] = quant_clip(xv / s_out) * s_out;
        float y = (xv / s_in) * wint[c + i] + bint[c + i];
        y *= sbn;
        y = fmaxf(y, 0.0f);
        int q = (int)quant_clip(y / s1);
        pack |= ((unsigned)(q & 0xff)) << (8 * i);
    }
    *(unsigned*)(q1 + (size_t)n * 256 + c) = pack;
}

__global__ __launch_bounds__(256) void gemm1(
    const signed char* __restrict__ q1, const signed char* __restrict__ w1,
    const float* __restrict__ b1f, const float* __restrict__ sc1,
    const float* __restrict__ s2p, signed char* __restrict__ q2) {
    __shared__ __align__(16) signed char As[128 * 256];
    __shared__ __align__(16) signed char Bs[128 * 256];
    int tid = threadIdx.x;
    int n0 = blockIdx.x * 128;

#if __has_builtin(__builtin_amdgcn_tensor_load_to_lds)
    if (tid == 0) {
        tdm_load_1d((unsigned)(uintptr_t)As, w1, 32768);
        tdm_load_1d((unsigned)(uintptr_t)Bs, q1 + (size_t)n0 * 256, 32768);
        tdm_wait();
    }
#else
    {
        const v4i* wsrc = (const v4i*)w1;
        v4i* adst = (v4i*)As;
        #pragma unroll
        for (int i = 0; i < 8; i++) adst[tid + i * 256] = wsrc[tid + i * 256];
        const v4i* xsrc = (const v4i*)(q1 + (size_t)n0 * 256);
        v4i* bdst = (v4i*)Bs;
        #pragma unroll
        for (int i = 0; i < 8; i++) bdst[tid + i * 256] = xsrc[tid + i * 256];
    }
#endif
    __builtin_prefetch(q1 + (size_t)(n0 + 128) * 256 + tid * 128, 0, 1);
    __syncthreads();

    int wave = tid >> 5, lane = tid & 31;
    int mw = (wave & 1) * 64, nw = (wave >> 1) * 32;
    int lr = lane & 15, lh = lane >> 4;

    v8i acc[4][2] = {};
    #pragma unroll
    for (int k = 0; k < 256; k += 64) {
        v8i a[4], bfr[2];
        #pragma unroll
        for (int mi = 0; mi < 4; mi++) {
            const signed char* ap = As + (size_t)(mw + mi * 16 + lr) * 256 + k + lh * 16;
            v2i a0 = *(const v2i*)(ap);
            v2i a1 = *(const v2i*)(ap + 8);
            v2i a2 = *(const v2i*)(ap + 32);
            v2i a3 = *(const v2i*)(ap + 40);
            v8i av; av[0]=a0[0]; av[1]=a0[1]; av[2]=a1[0]; av[3]=a1[1];
                    av[4]=a2[0]; av[5]=a2[1]; av[6]=a3[0]; av[7]=a3[1];
            a[mi] = av;
        }
        #pragma unroll
        for (int ni = 0; ni < 2; ni++) {
            const signed char* bp = Bs + (size_t)(nw + ni * 16 + lr) * 256 + k + lh * 16;
            v4i b0 = *(const v4i*)(bp);
            v4i b1 = *(const v4i*)(bp + 32);
            v8i bv; bv[0]=b0[0]; bv[1]=b0[1]; bv[2]=b0[2]; bv[3]=b0[3];
                    bv[4]=b1[0]; bv[5]=b1[1]; bv[6]=b1[2]; bv[7]=b1[3];
            bfr[ni] = bv;
        }
        #pragma unroll
        for (int mi = 0; mi < 4; mi++)
            #pragma unroll
            for (int ni = 0; ni < 2; ni++)
                acc[mi][ni] = __builtin_amdgcn_wmma_i32_16x16x64_iu8(
                    true, a[mi], true, bfr[ni], acc[mi][ni], false, false);
    }

    float s2 = s2p[0];
    #pragma unroll
    for (int mi = 0; mi < 4; mi++) {
        #pragma unroll
        for (int ni = 0; ni < 2; ni++) {
            int mbase = mw + mi * 16 + (lh ? 8 : 0);
            int n = n0 + nw + ni * 16 + lr;
            unsigned lo = 0, hi = 0;
            #pragma unroll
            for (int r = 0; r < 8; r++) {
                int m = mbase + r;
                float y = ((float)acc[mi][ni][r] + b1f[m]) * sc1[m];
                y = fmaxf(y, 0.0f);
                int q = (int)quant_clip(y / s2) & 0xff;
                if (r < 4) lo |= (unsigned)q << (8 * r);
                else       hi |= (unsigned)q << (8 * (r - 4));
            }
            v2i pk; pk[0] = (int)lo; pk[1] = (int)hi;
            *(v2i*)(q2 + (size_t)n * 128 + mbase) = pk;
        }
    }
}

__global__ __launch_bounds__(128) void gemm2(
    const signed char* __restrict__ q2, const signed char* __restrict__ w2,
    const float* __restrict__ sc2, const float* __restrict__ s_outp,
    float* __restrict__ out) {
    __shared__ __align__(16) signed char Ws[32 * 1152];
    int tid = threadIdx.x;

#if __has_builtin(__builtin_amdgcn_tensor_load_to_lds)
    if (tid == 0) {
        tdm_load_1d((unsigned)(uintptr_t)Ws, w2, 36864);
        tdm_wait();
    }
#else
    {
        const v4i* src = (const v4i*)w2;
        v4i* dst = (v4i*)Ws;
        #pragma unroll
        for (int i = 0; i < 18; i++) dst[tid + i * 128] = src[tid + i * 128];
    }
#endif
    __syncthreads();

    int wave = tid >> 5, lane = tid & 31;
    int lr = lane & 15, lh = lane >> 4;
    int img  = blockIdx.x / 49;
    int pblk = blockIdx.x - img * 49;
    int p = pblk * 64 + wave * 16 + lr;
    int y = p / 56, x = p - y * 56;
    const signed char* qimg = q2 + (size_t)img * 3136 * 128;

    v8i acc[2] = {};
    #pragma unroll
    for (int ky = 0; ky < 3; ky++) {
        #pragma unroll
        for (int kx = 0; kx < 3; kx++) {
            int sy = y + ky - 1, sx = x + kx - 1;
            bool inb = (sy >= 0) && (sy < 56) && (sx >= 0) && (sx < 56);
            const signed char* brow = qimg + (size_t)(sy * 56 + sx) * 128;
            int tap = ky * 3 + kx;
            #pragma unroll
            for (int kc = 0; kc < 2; kc++) {
                v4i b0 = {}, b1 = {};
                if (inb) {
                    const signed char* bb = brow + kc * 64 + lh * 16;
                    b0 = *(const v4i*)(bb);
                    b1 = *(const v4i*)(bb + 32);
                }
                v8i bv; bv[0]=b0[0]; bv[1]=b0[1]; bv[2]=b0[2]; bv[3]=b0[3];
                        bv[4]=b1[0]; bv[5]=b1[1]; bv[6]=b1[2]; bv[7]=b1[3];
                #pragma unroll
                for (int mi = 0; mi < 2; mi++) {
                    const signed char* ap = Ws + (size_t)(mi * 16 + lr) * 1152
                                               + tap * 128 + kc * 64 + lh * 16;
                    v2i a0 = *(const v2i*)(ap);
                    v2i a1 = *(const v2i*)(ap + 8);
                    v2i a2 = *(const v2i*)(ap + 32);
                    v2i a3 = *(const v2i*)(ap + 40);
                    v8i av; av[0]=a0[0]; av[1]=a0[1]; av[2]=a1[0]; av[3]=a1[1];
                            av[4]=a2[0]; av[5]=a2[1]; av[6]=a3[0]; av[7]=a3[1];
                    acc[mi] = __builtin_amdgcn_wmma_i32_16x16x64_iu8(
                        true, av, true, bv, acc[mi], false, false);
                }
            }
        }
    }

    float s_out = s_outp[0];
    #pragma unroll
    for (int mi = 0; mi < 2; mi++) {
        #pragma unroll
        for (int r = 0; r < 8; r++) {
            int m = mi * 16 + r + (lh ? 8 : 0);
            float v = (float)acc[mi][r] * sc2[m];
            float qv = quant_clip(v / s_out) * s_out;
            out[(size_t)img * 288 * 3136 + (size_t)(256 + m) * 3136 + p] = qv;
        }
    }
}

extern "C" void kernel_launch(void* const* d_in, const int* in_sizes, int n_in,
                              void* d_out, int out_size, void* d_ws, size_t ws_size,
                              hipStream_t stream) {
    (void)in_sizes; (void)n_in; (void)out_size; (void)ws_size;
    const float* batch = (const float*)d_in[0];
    const float* s_in  = (const float*)d_in[1];
    const float* g1    = (const float*)d_in[2];
    const float* b1    = (const float*)d_in[3];
    const float* m1    = (const float*)d_in[4];
    const float* v1    = (const float*)d_in[5];
    const float* w1f   = (const float*)d_in[6];
    const float* g2    = (const float*)d_in[7];
    const float* b2    = (const float*)d_in[8];
    const float* m2    = (const float*)d_in[9];
    const float* v2    = (const float*)d_in[10];
    const float* w2f   = (const float*)d_in[11];
    const float* s1    = (const float*)d_in[12];
    const float* s2    = (const float*)d_in[13];
    const float* s_out = (const float*)d_in[14];

    unsigned char* ws = (unsigned char*)d_ws;
    float* wbn  = (float*)(ws + OFF_WBN);
    float* bbn  = (float*)(ws + OFF_BBN);
    float* sbn  = (float*)(ws + OFF_SBN);
    signed char* w1q = (signed char*)(ws + OFF_W1);
    float* b1fq = (float*)(ws + OFF_B1);
    float* sc1  = (float*)(ws + OFF_SC1);
    signed char* w2q = (signed char*)(ws + OFF_W2);
    float* sc2  = (float*)(ws + OFF_SC2);
    signed char* q1  = (signed char*)(ws + OFF_Q1);
    signed char* q2  = (signed char*)(ws + OFF_Q2);

    float* out = (float*)d_out;
    float* out_tail = out + (size_t)32 * 288 * 3136;

    prep_bn<<<1, 256, 0, stream>>>(g1, b1, m1, v1, s_in, s_out, wbn, bbn, sbn, out_tail);
    prep_c1<<<128, 256, 0, stream>>>(w1f, g2, b2, m2, v2, s1, w1q, b1fq, sc1);
    prep_c2<<<32, 128, 0, stream>>>(w2f, s2, w2q, sc2);
    stage1<<<25088, 256, 0, stream>>>(batch, s_in, s1, s_out, wbn, bbn, sbn, q1, out);
    gemm1<<<784, 256, 0, stream>>>(q1, w1q, b1fq, sc1, s2, q2);
    gemm2<<<1568, 128, 0, stream>>>(q2, w2q, sc2, s_out, out);
}
